// IPEXTransformerAtten_76751065579540
// MI455X (gfx1250) — matrix operand order repaired and checked by reading, not compiled, not run on hardware
//
#include <hip/hip_runtime.h>
#include <stdint.h>

typedef __attribute__((ext_vector_type(16))) __bf16 v16bf;
typedef __attribute__((ext_vector_type(8)))  float  v8f;
typedef __attribute__((ext_vector_type(4)))  unsigned int u32x4;

union AFrag { v16bf v; u32x4 u[2]; };

constexpr int SEQ = 2048, BATCH = 2, EMB = 2048, NH = 32, HD = 64;
constexpr int MROWS = SEQ * BATCH;   // 4096
constexpr int N_QKV = 3 * EMB;       // 6144
constexpr float SCALE    = 0.125f;   // 1/sqrt(64)
constexpr float MASK_NEG = -66504.0f;

__device__ __forceinline__ __bf16 f2bf(float f)  { return (__bf16)f; }
__device__ __forceinline__ float  bf2f(__bf16 b) { return (float)b; }
__device__ __forceinline__ void store_val(float* p, float v)  { *p = v; }
__device__ __forceinline__ void store_val(__bf16* p, float v) { *p = f2bf(v); }

__device__ __forceinline__ v8f wmma_bf16(const AFrag& a, const AFrag& b, v8f c) {
  return __builtin_amdgcn_wmma_f32_16x16x32_bf16(false, a.v, false, b.v,
                                                 (short)0, c, false, false);
}

// ---------------------------------------------------------------- convert f32->bf16
__global__ void convert_f32_bf16(const float* __restrict__ in,
                                 __bf16* __restrict__ out, int n) {
  int i = blockIdx.x * blockDim.x + threadIdx.x;
  if (i < n) out[i] = f2bf(in[i]);
}

// ------------------------------------------------- transpose [R,C] f32 -> [C,R] bf16
__global__ void __launch_bounds__(256)
transpose_f32_to_bf16(const float* __restrict__ in, __bf16* __restrict__ out,
                      int R, int C) {
  __shared__ float tile[32][33];
  const int c0 = blockIdx.x * 32, r0 = blockIdx.y * 32;
  #pragma unroll
  for (int k = 0; k < 32; k += 8)
    tile[threadIdx.y + k][threadIdx.x] =
        in[(size_t)(r0 + threadIdx.y + k) * C + c0 + threadIdx.x];
  __syncthreads();
  #pragma unroll
  for (int k = 0; k < 32; k += 8)
    out[(size_t)(c0 + threadIdx.y + k) * R + r0 + threadIdx.x] =
        f2bf(tile[threadIdx.x][threadIdx.y + k]);
}

// ------------------------------------------------------------- WMMA GEMM + bias
// C[M,N] = A[M,K](bf16,row-major) * Bt[N,K](bf16, = B transposed) + bias[N]
// Block: 256 thr (8 waves), tile 128x256x32; wave grid 2x4 -> 64x64 per wave
// (16 WMMAs / wave / k-step). Global->LDS double-buffered through registers.
template <typename OutT>
__global__ void __launch_bounds__(256)
gemm_bf16_wmma(const __bf16* __restrict__ A, const __bf16* __restrict__ Bt,
               const float* __restrict__ bias, OutT* __restrict__ out,
               int M, int N, int K) {
  __shared__ __align__(16) __bf16 As[128 * 40];  // stride 40: conflict-free b128 reads
  __shared__ __align__(16) __bf16 Bs[256 * 40];
  const int tid  = threadIdx.x;
  const int lane = tid & 31;
  const int l16  = lane & 15;
  const bool hi  = lane >= 16;
  const int wid  = tid >> 5;
  const int wm   = wid & 1;    // 64-row slice
  const int wn   = wid >> 1;   // 64-col slice
  const int bm   = blockIdx.y * 128;
  const int bn   = blockIdx.x * 256;

  const int ra = tid >> 2;         // 0..63 : tile row per load pass
  const int ca = (tid & 3) * 8;    // 0,8,16,24 : k-offset per load pass

  v8f acc[4][4] = {};

  // initial tile (k0 = 0) -> LDS
  #pragma unroll
  for (int p = 0; p < 2; ++p)
    *(u32x4*)&As[(p * 64 + ra) * 40 + ca] =
        *(const u32x4*)(A + (size_t)(bm + p * 64 + ra) * K + ca);
  #pragma unroll
  for (int p = 0; p < 4; ++p)
    *(u32x4*)&Bs[(p * 64 + ra) * 40 + ca] =
        *(const u32x4*)(Bt + (size_t)(bn + p * 64 + ra) * K + ca);
  __syncthreads();

  for (int k0 = 0; k0 < K; k0 += 32) {
    const bool more = (k0 + 32) < K;
    u32x4 pa[2], pbv[4];
    if (more) {                      // prefetch next k-slab into registers
      #pragma unroll
      for (int p = 0; p < 2; ++p)
        pa[p] = *(const u32x4*)(A + (size_t)(bm + p * 64 + ra) * K + k0 + 32 + ca);
      #pragma unroll
      for (int p = 0; p < 4; ++p)
        pbv[p] = *(const u32x4*)(Bt + (size_t)(bn + p * 64 + ra) * K + k0 + 32 + ca);
    }

    AFrag a[4];
    #pragma unroll
    for (int i = 0; i < 4; ++i) {    // A layout: lo K0..15, hi K16..31
      const __bf16* ap = &As[(wm * 64 + i * 16 + l16) * 40];
      a[i].u[0] = *(const u32x4*)(ap + (hi ? 16 : 0));
      a[i].u[1] = *(const u32x4*)(ap + (hi ? 16 : 0) + 8);
    }
    #pragma unroll
    for (int j = 0; j < 4; ++j) {    // B layout: lo K0..15, hi K16..31
      const __bf16* bp = &Bs[(wn * 64 + j * 16 + l16) * 40];
      AFrag bf;
      bf.u[0] = *(const u32x4*)(bp + (hi ? 16 : 0));
      bf.u[1] = *(const u32x4*)(bp + (hi ? 16 : 0) + 8);
      #pragma unroll
      for (int i = 0; i < 4; ++i)
        acc[i][j] = wmma_bf16(a[i], bf, acc[i][j]);
    }

    __syncthreads();
    if (more) {
      #pragma unroll
      for (int p = 0; p < 2; ++p)
        *(u32x4*)&As[(p * 64 + ra) * 40 + ca] = pa[p];
      #pragma unroll
      for (int p = 0; p < 4; ++p)
        *(u32x4*)&Bs[(p * 64 + ra) * 40 + ca] = pbv[p];
      __syncthreads();
    }
  }

  #pragma unroll
  for (int i = 0; i < 4; ++i)
    #pragma unroll
    for (int j = 0; j < 4; ++j) {
      const int col = bn + wn * 64 + j * 16 + l16;
      const float bv = bias[col];
      #pragma unroll
      for (int r = 0; r < 8; ++r) {
        const int row = bm + wm * 64 + i * 16 + (hi ? 8 : 0) + r;
        store_val(out + (size_t)row * N + col, acc[i][j][r] + bv);
      }
    }
}

// --------------------------------------- RoPE + scatter q,k:[B,H,S,D]  v:[B,H,D,S]
__global__ void __launch_bounds__(256)
rope_scatter_kernel(const __bf16* __restrict__ qkv,
                    __bf16* __restrict__ qh, __bf16* __restrict__ kh,
                    __bf16* __restrict__ vt) {
  const int t  = blockIdx.x * blockDim.x + threadIdx.x;  // S*B*H*(D/2)
  const int d2 = t & 31;
  const int h  = (t >> 5) & 31;
  const int b  = (t >> 10) & 1;
  const int s  = t >> 11;
  if (s >= SEQ) return;
  const size_t base = (size_t)(s * BATCH + b) * N_QKV + h * HD;
  const float q0 = bf2f(qkv[base + d2]);
  const float q1 = bf2f(qkv[base + d2 + 32]);
  const float k0 = bf2f(qkv[base + EMB + d2]);
  const float k1 = bf2f(qkv[base + EMB + d2 + 32]);
  const float v0 = bf2f(qkv[base + 2 * EMB + d2]);
  const float v1 = bf2f(qkv[base + 2 * EMB + d2 + 32]);
  const float inv = __powf(10000.0f, -(float)d2 * (1.0f / 32.0f));
  const float th  = (float)s * inv;
  const float cs = cosf(th), sn = sinf(th);
  const size_t bh = (size_t)(b * NH + h);
  __bf16* qo = qh + (bh * SEQ + s) * HD;
  __bf16* ko = kh + (bh * SEQ + s) * HD;
  qo[d2]      = f2bf(q0 * cs - q1 * sn);
  qo[d2 + 32] = f2bf(q1 * cs + q0 * sn);
  ko[d2]      = f2bf(k0 * cs - k1 * sn);
  ko[d2 + 32] = f2bf(k1 * cs + k0 * sn);
  __bf16* vo = vt + bh * HD * SEQ + s;
  vo[(size_t)d2 * SEQ]        = f2bf(v0);
  vo[(size_t)(d2 + 32) * SEQ] = f2bf(v1);
}

// ------------------------------------------------------------- flash attention
// grid (S/128, B*H), block 256. Each wave: 16 query rows, loop 32 keys/step.
// Tile-wide running max (5 shuffles/chunk), deferred row-sum reduction.
__global__ void __launch_bounds__(256)
flash_attn_kernel(const __bf16* __restrict__ qh, const __bf16* __restrict__ kh,
                  const __bf16* __restrict__ vt, __bf16* __restrict__ ctx) {
  __shared__ __align__(16) __bf16 pbuf[8 * 16 * 40];
  const int tid  = threadIdx.x;
  const int w    = tid >> 5;
  const int lane = tid & 31;
  const int l16  = lane & 15;
  const bool hi  = lane >= 16;
  const int bh   = blockIdx.y;
  const int b    = bh / NH;
  const int h    = bh % NH;
  const int qs   = blockIdx.x * 128 + w * 16;

  const __bf16* qbase = qh + (size_t)bh * SEQ * HD;
  const __bf16* kbase = kh + (size_t)bh * SEQ * HD;
  const __bf16* vbase = vt + (size_t)bh * HD * SEQ;

  AFrag aQ[2];
  {
    const __bf16* qrow = qbase + (size_t)(qs + l16) * HD;
    #pragma unroll
    for (int c = 0; c < 2; ++c) {
      aQ[c].u[0] = *(const u32x4*)(qrow + c * 32 + (hi ? 16 : 0));
      aQ[c].u[1] = *(const u32x4*)(qrow + c * 32 + (hi ? 16 : 0) + 8);
    }
  }

  v8f acc[4] = {};
  float lsum[8];
  #pragma unroll
  for (int r = 0; r < 8; ++r) lsum[r] = 0.0f;
  float m_run = -3.0e38f;

  __bf16* pb = pbuf + w * (16 * 40);
  const int q_hi = qs + 15;

  for (int j0 = 0; j0 <= q_hi; j0 += 32) {
    // ---- scores: two 16x16 key tiles, K-dim = head dim (2 x 32) ----
    v8f s0 = {}, s1 = {};
    {
      const __bf16* kr0 = kbase + (size_t)(j0 + l16) * HD;
      const __bf16* kr1 = kr0 + 16 * HD;
      #pragma unroll
      for (int c = 0; c < 2; ++c) {
        AFrag bk0, bk1;
        bk0.u[0] = *(const u32x4*)(kr0 + c * 32 + (hi ? 16 : 0));
        bk0.u[1] = *(const u32x4*)(kr0 + c * 32 + (hi ? 16 : 0) + 8);
        bk1.u[0] = *(const u32x4*)(kr1 + c * 32 + (hi ? 16 : 0));
        bk1.u[1] = *(const u32x4*)(kr1 + c * 32 + (hi ? 16 : 0) + 8);
        s0 = wmma_bf16(aQ[c], bk0, s0);
        s1 = wmma_bf16(aQ[c], bk1, s1);
      }
    }

    // ---- scale + (diagonal chunks only) causal mask; tile-wide max ----
    const bool edge = (j0 + 31) > qs;   // wave-uniform
    float vv0[8], vv1[8];
    float mx = -3.0e38f;
    #pragma unroll
    for (int r = 0; r < 8; ++r) {
      float v0 = s0[r] * SCALE;
      float v1 = s1[r] * SCALE;
      if (edge) {
        const int qrow = qs + (hi ? 8 : 0) + r;
        const int kpos = j0 + l16;
        v0 += (kpos      > qrow) ? MASK_NEG : 0.0f;
        v1 += (kpos + 16 > qrow) ? MASK_NEG : 0.0f;
      }
      vv0[r] = v0; vv1[r] = v1;
      mx = fmaxf(mx, fmaxf(v0, v1));
    }
    #pragma unroll
    for (int o = 16; o >= 1; o >>= 1) mx = fmaxf(mx, __shfl_xor(mx, o, 32));
    const float mnew  = fmaxf(m_run, mx);
    const float alpha = __expf(m_run - mnew);
    m_run = mnew;
    #pragma unroll
    for (int t = 0; t < 4; ++t) acc[t] *= alpha;

    // ---- exp, per-lane partial row sums, P -> LDS (bf16, A layout target) ----
    #pragma unroll
    for (int r = 0; r < 8; ++r) {
      const float p0 = __expf(vv0[r] - mnew);
      const float p1 = __expf(vv1[r] - mnew);
      lsum[r] = lsum[r] * alpha + p0 + p1;
      const int row = (hi ? 8 : 0) + r;
      pb[row * 40 + l16]      = f2bf(p0);
      pb[row * 40 + 16 + l16] = f2bf(p1);
    }
    asm volatile("s_wait_dscnt 0" ::: "memory");  // wave-local LDS RAW fence

    AFrag aP;                                     // P in A layout
    {
      const __bf16* pr = pb + l16 * 40;
      aP.u[0] = *(const u32x4*)(pr + (hi ? 16 : 0));
      aP.u[1] = *(const u32x4*)(pr + (hi ? 16 : 0) + 8);
    }

    // ---- acc += P x V   (V stored [d][s]: B-columns contiguous) ----
    #pragma unroll
    for (int t = 0; t < 4; ++t) {
      const __bf16* vr = vbase + (size_t)(t * 16 + l16) * SEQ + j0 + (hi ? 16 : 0);
      AFrag bv;
      bv.u[0] = *(const u32x4*)(vr);
      bv.u[1] = *(const u32x4*)(vr + 8);
      acc[t] = wmma_bf16(aP, bv, acc[t]);
    }
  }

  // ---- one deferred cross-lane row-sum reduction + writeback ----
  #pragma unroll
  for (int r = 0; r < 8; ++r) {
    float ls = lsum[r];
    #pragma unroll
    for (int o = 8; o >= 1; o >>= 1) ls += __shfl_xor(ls, o, 32);
    const float il = 1.0f / ls;
    const int s_pos = qs + (hi ? 8 : 0) + r;
    __bf16* crow = ctx + (size_t)(s_pos * BATCH + b) * EMB + h * HD;
    #pragma unroll
    for (int t = 0; t < 4; ++t)
      crow[t * 16 + l16] = f2bf(acc[t][r] * il);
  }
}

// ---------------------------------------------------------------------- launch
extern "C" void kernel_launch(void* const* d_in, const int* in_sizes, int n_in,
                              void* d_out, int out_size, void* d_ws, size_t ws_size,
                              hipStream_t stream) {
  (void)in_sizes; (void)n_in; (void)out_size; (void)ws_size;
  const float* hid   = (const float*)d_in[0];
  const float* qkv_w = (const float*)d_in[1];
  const float* qkv_b = (const float*)d_in[2];
  const float* out_w = (const float*)d_in[3];
  const float* out_b = (const float*)d_in[4];

  char* ws = (char*)d_ws;
  size_t off = 0;
  __bf16* hidb  = (__bf16*)(ws + off); off += (size_t)MROWS * EMB * 2;
  __bf16* wqkvT = (__bf16*)(ws + off); off += (size_t)N_QKV * EMB * 2;
  __bf16* woutT = (__bf16*)(ws + off); off += (size_t)EMB * EMB * 2;
  __bf16* qkvb  = (__bf16*)(ws + off); off += (size_t)MROWS * N_QKV * 2;
  __bf16* qhp   = (__bf16*)(ws + off); off += (size_t)BATCH * NH * SEQ * HD * 2;
  __bf16* khp   = (__bf16*)(ws + off); off += (size_t)BATCH * NH * SEQ * HD * 2;
  __bf16* vtp   = (__bf16*)(ws + off); off += (size_t)BATCH * NH * SEQ * HD * 2;
  __bf16* ctx   = (__bf16*)(ws + off);

  convert_f32_bf16<<<(MROWS * EMB) / 256, 256, 0, stream>>>(hid, hidb, MROWS * EMB);
  transpose_f32_to_bf16<<<dim3(N_QKV / 32, EMB / 32), dim3(32, 8), 0, stream>>>(
      qkv_w, wqkvT, EMB, N_QKV);
  transpose_f32_to_bf16<<<dim3(EMB / 32, EMB / 32), dim3(32, 8), 0, stream>>>(
      out_w, woutT, EMB, EMB);
  gemm_bf16_wmma<__bf16><<<dim3(N_QKV / 256, MROWS / 128), 256, 0, stream>>>(
      hidb, wqkvT, qkv_b, qkvb, MROWS, N_QKV, EMB);
  rope_scatter_kernel<<<(SEQ * BATCH * NH * 32) / 256, 256, 0, stream>>>(
      qkvb, qhp, khp, vtp);
  flash_attn_kernel<<<dim3(SEQ / 128, BATCH * NH), 256, 0, stream>>>(
      qhp, khp, vtp, ctx);
  gemm_bf16_wmma<float><<<dim3(EMB / 256, MROWS / 128), 256, 0, stream>>>(
      ctx, woutT, out_b, (float*)d_out, MROWS, EMB, EMB);
}
